// DBLSTM_32882269618468
// MI455X (gfx1250) — matrix operand order repaired and checked by reading, not compiled, not run on hardware
//
#include <hip/hip_runtime.h>

// ---------------------------------------------------------------------------
// DBLSTM for MI455X (gfx1250): bf16 WMMA GEMMs + persistent-workgroup scans,
// with Tensor-Data-Mover (TDM) staging of the per-step gate rows in the
// latency-critical layer-1 recurrence.
// B=32, T=1024, D=H=256, gates N = 4H = 1024.
// ---------------------------------------------------------------------------

#define T_   1024
#define B_   32
#define D_   256
#define H_   256
#define NG   1024            // 4*H
#define TBROWS (T_ * B_)     // 32768 rows of the projection GEMMs
#define GLD  1028            // padded LDS row (floats) for the gates buffer
#define HPAD 272             // padded LDS row (bf16) for the h buffer
#define HULD 260             // padded LDS row (floats) for the hu buffer

typedef __attribute__((ext_vector_type(8)))  unsigned int v8u;
typedef __attribute__((ext_vector_type(8)))  float        v8f;
typedef __attribute__((ext_vector_type(16))) __bf16       v16bf;
typedef __attribute__((ext_vector_type(4)))  unsigned int u32x4;
typedef __attribute__((ext_vector_type(8)))  int          i32x8;
typedef __attribute__((ext_vector_type(4)))  int          i32x4;

#if defined(__gfx1250__) && __has_builtin(__builtin_amdgcn_tensor_load_to_lds)
#define HAS_TDM 1
#else
#define HAS_TDM 0
#endif

__device__ inline v16bf bc16(v8u x) { return __builtin_bit_cast(v16bf, x); }

__device__ inline v8f wmma_bf16(v8u a, v8u b, v8f c) {
  // D = A(16x32 bf16) * B(32x16 bf16) + C(16x16 f32)
  return __builtin_amdgcn_wmma_f32_16x16x32_bf16(false, bc16(a), false, bc16(b),
                                                 (short)0, c, false, false);
}

__device__ inline unsigned short f2bf(float f) {
  unsigned int u = __float_as_uint(f);
  u += 0x7FFFu + ((u >> 16) & 1u);      // round-to-nearest-even
  return (unsigned short)(u >> 16);
}

__device__ inline float sigm(float x) { return 1.0f / (1.0f + __expf(-x)); }

// A fragment (16x32 bf16, row-major source). ISA layout: lanes 0-15 hold
// rows M=0..15; VGPR j holds K pairs {2j,2j+1} (j<4) / {2j+8,..} (j>=4),
// lanes 16-31 shift K by +8.
__device__ inline v8u load_a_frag(const unsigned short* base, int ldk) {
  int lane = threadIdx.x & 31;
  int m = lane & 15, hl = lane >> 4;
  const unsigned short* p = base + (size_t)m * ldk + 8 * hl;
  v8u r;
#pragma unroll
  for (int j = 0; j < 8; ++j) {
    int k = 2 * j + ((j >= 4) ? 8 : 0);
    r[j] = *(const unsigned int*)(p + k);
  }
  return r;
}

// B fragment (32x16 bf16) from (n,k)-major weights. Same K order as A: VGPR j
// holds K pairs {2j,2j+1} (j<4) / {2j+8,..} (j>=4), lanes 16-31 shift K by +8.
__device__ inline v8u load_b_frag(const unsigned short* base, int ldk) {
  int lane = threadIdx.x & 31;
  int n = lane & 15, kh = lane >> 4;
  const unsigned short* p = base + (size_t)n * ldk + 8 * kh;
  v8u r;
#pragma unroll
  for (int j = 0; j < 8; ++j) r[j] = *(const unsigned int*)(p + 2 * j + ((j >= 4) ? 8 : 0));
  return r;
}

#if HAS_TDM
// Issue a TDM DMA of one (B_ x NG) f32 row of xp into LDS.
// D# group0: count=1, lds_addr, global_addr, type=2("image").
// D# group1: data_size=4B, tensor_dim0=tile_dim0=32768, tensor_dim1=tile_dim1=1,
//            tensor_dim0_stride=32768.
__device__ inline void tdm_stage_row(const float* gsrc, void* lds_dst) {
  unsigned long long ga = (unsigned long long)(uintptr_t)gsrc;
  unsigned lds = (unsigned)(uintptr_t)lds_dst;   // LDS aperture: addr[31:0]
  u32x4 g0;
  g0[0] = 1u;                                    // count=1 valid descriptor
  g0[1] = lds;                                   // lds_addr
  g0[2] = (unsigned)(ga & 0xFFFFFFFFu);          // global_addr[31:0]
  g0[3] = (unsigned)((ga >> 32) & 0x01FFFFFFu) | 0x80000000u; // addr[56:32]|type=2
  i32x8 g1;
  g1[0] = (int)0x00020000;   // wg_mask=0, data_size=2 (4 bytes)
  g1[1] = (int)0x80000000;   // tensor_dim0[15:0]=0x8000 in bits[31:16]
  g1[2] = (int)0x00010000;   // tensor_dim0[31:16]=0, tensor_dim1[15:0]=1
  g1[3] = (int)0x80000000;   // tile_dim0=0x8000 in bits[31:16]
  g1[4] = (int)0x00000001;   // tile_dim1=1, tile_dim2=0
  g1[5] = (int)32768;        // tensor_dim0_stride[31:0]
  g1[6] = 0;
  g1[7] = 0;
  i32x4 z4 = {0, 0, 0, 0};
#if __has_include(<hip/amd_detail/amd_gfx1250_TDM.h>)
  i32x8 z8 = {0, 0, 0, 0, 0, 0, 0, 0};
  __builtin_amdgcn_tensor_load_to_lds(g0, g1, z4, z4, z8, 0);
#else
  __builtin_amdgcn_tensor_load_to_lds(g0, g1, z4, z4, 0);
#endif
}
#endif

// ------------------------- layout conversion kernels -----------------------

// x (B,T,D) f32 -> xT (T,B,D) bf16
__global__ __launch_bounds__(256) void k_conv_x(const float* __restrict__ x,
                                                unsigned short* __restrict__ xT) {
  size_t i = (size_t)blockIdx.x * 256 + threadIdx.x;     // (t,b,d) flat
  int d = (int)(i & 255);
  size_t tb = i >> 8;
  int b = (int)(tb & 31);
  int t = (int)(tb >> 5);
  xT[i] = f2bf(x[((size_t)b * T_ + t) * D_ + d]);
}

// W (4,K,H) f32 -> Wt (N=4H, K) bf16 with Wt[(g*H+h)*K + k] = W[g][k][h]
__global__ __launch_bounds__(256) void k_conv_w(const float* __restrict__ W,
                                                unsigned short* __restrict__ Wt,
                                                int K) {
  size_t i = (size_t)blockIdx.x * 256 + threadIdx.x;     // c*K + k
  int k = (int)(i % K);
  size_t c = i / K;
  int g = (int)(c >> 8), h = (int)(c & 255);
  Wt[i] = f2bf(W[((size_t)g * K + k) * H_ + h]);
}

// Stacked layer-2 weights: Wt[(g*H+h)*512 + k] = k<256 ? Wx[g][k][h] : Wh[g][k-256][h]
__global__ __launch_bounds__(256) void k_conv_w2(const float* __restrict__ Wx,
                                                 const float* __restrict__ Wh,
                                                 unsigned short* __restrict__ Wt) {
  size_t i = (size_t)blockIdx.x * 256 + threadIdx.x;     // c*512 + k
  int k = (int)(i & 511);
  size_t c = i >> 9;
  int g = (int)(c >> 8), h = (int)(c & 255);
  float v = (k < 256) ? Wx[((size_t)g * 256 + k) * 256 + h]
                      : Wh[((size_t)g * 256 + (k - 256)) * 256 + h];
  Wt[i] = f2bf(v);
}

// U2t[hc*256 + k] = U[0][k][hc]
__global__ __launch_bounds__(256) void k_conv_u2(const float* __restrict__ U,
                                                 unsigned short* __restrict__ Ut) {
  size_t i = (size_t)blockIdx.x * 256 + threadIdx.x;     // hc*256 + k
  int k = (int)(i & 255);
  int hc = (int)(i >> 8);
  Ut[i] = f2bf(U[(size_t)k * 256 + hc]);
}

// ------------------------------ projection GEMM ----------------------------
// Cout[(t*B+b), c] = sum_k A(t,b,k) * Wt[c,k]  (+ bias[c])
// A = [A0 (k<K0, optionally time-reversed) ; A1 (k>=K0, scan order)]
// Block: 256 threads = 8 waves, 128x128 tile; wave tile 32x64 (2m x 4n).
__global__ __launch_bounds__(256) void k_gemm(const unsigned short* __restrict__ A0,
                                              const unsigned short* __restrict__ A1,
                                              const unsigned short* __restrict__ Wt,
                                              const float* __restrict__ bias,
                                              float* __restrict__ Cout,
                                              int K0, int Ktot, int revA0) {
  int wave = threadIdx.x >> 5;
  int lane = threadIdx.x & 31;
  int mw = wave >> 1, nw = wave & 1;
  int row0 = blockIdx.x * 128 + mw * 32;       // multiple of 32 -> single t
  int col0 = blockIdx.y * 128 + nw * 64;
  int t  = row0 >> 5;
  int tA = revA0 ? (T_ - 1 - t) : t;

  v8f zero8 = {0.f, 0.f, 0.f, 0.f, 0.f, 0.f, 0.f, 0.f};
  v8f acc[2][4];
#pragma unroll
  for (int i = 0; i < 2; ++i)
#pragma unroll
    for (int j = 0; j < 4; ++j) acc[i][j] = zero8;

  for (int kt = 0; kt < Ktot; kt += 32) {
    const unsigned short* Ab;
    int lda;
    if (kt < K0) { Ab = A0 + ((size_t)tA * B_) * K0 + kt;        lda = K0; }
    else         { Ab = A1 + ((size_t)t  * B_) * H_ + (kt - K0); lda = H_; }
    v8u a0 = load_a_frag(Ab, lda);
    v8u a1 = load_a_frag(Ab + (size_t)16 * lda, lda);
#pragma unroll
    for (int nt = 0; nt < 4; ++nt) {
      v8u bfr = load_b_frag(Wt + (size_t)(col0 + nt * 16) * Ktot + kt, Ktot);
      acc[0][nt] = wmma_bf16(a0, bfr, acc[0][nt]);
      acc[1][nt] = wmma_bf16(a1, bfr, acc[1][nt]);
    }
  }

  int nn = lane & 15, hlf = lane >> 4;
#pragma unroll
  for (int mt = 0; mt < 2; ++mt)
#pragma unroll
    for (int nt = 0; nt < 4; ++nt)
#pragma unroll
      for (int v = 0; v < 8; ++v) {
        int r = row0 + mt * 16 + v + 8 * hlf;     // C layout: M = v + 8*half
        int c = col0 + nt * 16 + nn;              //           N = lane&15
        float val = acc[mt][nt][v];
        if (bias) val += bias[c];
        Cout[(size_t)r * NG + c] = val;
      }
}

// ------------------------------ layer-1 scan -------------------------------
// grid = 2 (fwd/bwd), 1024 threads = 32 waves. Per step: gates(32x1024) =
// sigmoid(xp_t + h*U) via WMMA (wave owns 32 cols), then c/h update; c in
// registers for all 1024 steps, h exchanged through LDS in bf16. The next
// step's xp row (128KB) is DMA'd into LDS by the Tensor Data Mover while the
// current step's state update runs, fenced with s_wait_tensorcnt + barrier.
__global__ __launch_bounds__(1024) void k_scan1(const float* __restrict__ xpf,
                                                const float* __restrict__ xpb,
                                                const unsigned short* __restrict__ U1f,
                                                const unsigned short* __restrict__ U1b,
                                                unsigned short* __restrict__ hpf,
                                                unsigned short* __restrict__ hpb) {
  int dir = blockIdx.x;
  const float* xp = dir ? xpb : xpf;
  const unsigned short* U = dir ? U1b : U1f;
  unsigned short* hp = dir ? hpb : hpf;

  extern __shared__ char smem[];
  float* gatesS = (float*)smem;                                   // 32 x GLD f32
  unsigned short* hA = (unsigned short*)(smem + (size_t)32 * GLD * 4); // 32 x HPAD
  float* hXP = (float*)(smem + (size_t)32 * GLD * 4 + (size_t)32 * HPAD * 2); // 32 x NG f32

  int tid = threadIdx.x;
  int wave = tid >> 5, lane = tid & 31;
  int n0 = wave * 32;
  int nn = lane & 15, hlf = lane >> 4;

  for (int i = tid; i < 32 * HPAD; i += 1024) hA[i] = 0;
  float cst[8];
#pragma unroll
  for (int j = 0; j < 8; ++j) cst[j] = 0.0f;

#if HAS_TDM
  if (wave == 0) tdm_stage_row(xp, hXP);        // stage xp_0
#endif

  for (int t = 0; t < T_; ++t) {
#if HAS_TDM
    __builtin_amdgcn_s_wait_tensorcnt(0);       // wave0 waits; others no-op
    __syncthreads();                            // xp_t staged + hA visible
    const float* xprow = hXP;
#else
    __syncthreads();
    const float* xprow = xp + (size_t)t * (B_ * NG);
    if (t + 1 < T_)
      __builtin_prefetch(xp + (size_t)(t + 1) * (B_ * NG) + (size_t)tid * 32, 0, 1);
#endif

#pragma unroll
    for (int nt = 0; nt < 2; ++nt) {
      int c0 = n0 + nt * 16;
      const unsigned short* Ub = U + (size_t)c0 * H_;
      v8f acc0 = {0.f, 0.f, 0.f, 0.f, 0.f, 0.f, 0.f, 0.f};
      v8f acc1 = acc0;
#pragma unroll
      for (int ki = 0; ki < 8; ++ki) {
        v8u bfr = load_b_frag(Ub + ki * 32, H_);              // L2-resident U
        v8u a0  = load_a_frag(hA + ki * 32, HPAD);            // LDS h, rows 0-15
        v8u a1  = load_a_frag(hA + 16 * HPAD + ki * 32, HPAD);// rows 16-31
        acc0 = wmma_bf16(a0, bfr, acc0);
        acc1 = wmma_bf16(a1, bfr, acc1);
      }
#pragma unroll
      for (int v = 0; v < 8; ++v) {
        int r0 = v + 8 * hlf;
        gatesS[r0 * GLD + c0 + nn] =
            sigm(acc0[v] + xprow[(size_t)r0 * NG + c0 + nn]);
        gatesS[(16 + r0) * GLD + c0 + nn] =
            sigm(acc1[v] + xprow[(size_t)(16 + r0) * NG + c0 + nn]);
      }
    }
    __syncthreads();                            // all hXP/hA readers done

#if HAS_TDM
    if (wave == 0 && t + 1 < T_)                // DMA next row, overlapped
      tdm_stage_row(xp + (size_t)(t + 1) * (B_ * NG), hXP);
#endif

#pragma unroll
    for (int j = 0; j < 8; ++j) {
      int flat = tid + 1024 * j;          // (b, hcol) over 32x256 states
      int b = flat >> 8, hc = flat & 255;
      float gi = gatesS[b * GLD + hc];
      float gf = gatesS[b * GLD + 256 + hc];
      float gg = gatesS[b * GLD + 512 + hc];
      float go = gatesS[b * GLD + 768 + hc];
      float cn = gf + cst[j] + gi * gg;   // reference: c = f + c + i*g
      cst[j] = cn;
      float h = go + tanhf(cn);           // reference: h = o + tanh(c)
      unsigned short hb = f2bf(h);
      hA[b * HPAD + hc] = hb;
      hp[(size_t)t * (B_ * H_) + b * H_ + hc] = hb;   // scan-order bf16 h
    }
  }
}

// ------------------------------ layer-2 scan -------------------------------
// grid = 2 (fwd/bwd), 512 threads = 16 waves. hu(32x256) = h*U[0] with U
// resident in VGPRs (8 B-fragments/wave); gates broadcast hu across 4 gates.
__global__ __launch_bounds__(512) void k_scan2(const float* __restrict__ xpf,
                                               const float* __restrict__ xpb,
                                               const unsigned short* __restrict__ U2f,
                                               const unsigned short* __restrict__ U2b,
                                               const float* __restrict__ b2f,
                                               const float* __restrict__ b2b,
                                               float* __restrict__ out_t,
                                               float* __restrict__ out_net) {
  int dir = blockIdx.x;
  const float* xp = dir ? xpb : xpf;
  const unsigned short* U = dir ? U2b : U2f;
  const float* bias = dir ? b2b : b2f;

  __shared__ float huS[32 * HULD];
  __shared__ unsigned short hA2[32 * HPAD];

  int tid = threadIdx.x;
  int wave = tid >> 5, lane = tid & 31;
  int c0 = wave * 16;
  int nn = lane & 15, hlf = lane >> 4;

  v8u bU[8];                                   // persistent U fragments
#pragma unroll
  for (int ki = 0; ki < 8; ++ki)
    bU[ki] = load_b_frag(U + (size_t)c0 * H_ + ki * 32, H_);

  for (int i = tid; i < 32 * HPAD; i += 512) hA2[i] = 0;
  float cst[16];
#pragma unroll
  for (int j = 0; j < 16; ++j) cst[j] = 0.0f;
  __syncthreads();

  for (int t = 0; t < T_; ++t) {
    if (t + 1 < T_)
      __builtin_prefetch(xp + (size_t)(t + 1) * (B_ * NG) + (size_t)tid * 64, 0, 1);

    v8f acc0 = {0.f, 0.f, 0.f, 0.f, 0.f, 0.f, 0.f, 0.f};
    v8f acc1 = acc0;
#pragma unroll
    for (int ki = 0; ki < 8; ++ki) {
      v8u a0 = load_a_frag(hA2 + ki * 32, HPAD);
      v8u a1 = load_a_frag(hA2 + 16 * HPAD + ki * 32, HPAD);
      acc0 = wmma_bf16(a0, bU[ki], acc0);
      acc1 = wmma_bf16(a1, bU[ki], acc1);
    }
#pragma unroll
    for (int v = 0; v < 8; ++v) {
      huS[(v + 8 * hlf) * HULD + c0 + nn] = acc0[v];
      huS[(16 + v + 8 * hlf) * HULD + c0 + nn] = acc1[v];
    }
    __syncthreads();

    int treal = dir ? (T_ - 1 - t) : t;
#pragma unroll
    for (int j = 0; j < 16; ++j) {
      int flat = tid + 512 * j;
      int b = flat >> 8, hc = flat & 255;
      float huv = huS[b * HULD + hc] + bias[hc];        // hu = h@U[0] + b[0]
      const float* xr = xp + ((size_t)t * B_ + b) * NG;
      float gi = sigm(xr[hc] + huv);
      float gf = sigm(xr[256 + hc] + huv);
      float gg = sigm(xr[512 + hc] + huv);
      float go = sigm(xr[768 + hc] + huv);
      float cn = gf + cst[j] + gi * gg;
      cst[j] = cn;
      float h = go + tanhf(cn);
      hA2[b * HPAD + hc] = f2bf(h);
      out_net[((size_t)b * T_ + treal) * (2 * H_) + dir * H_ + hc] = h;
      if (t == T_ - 1) out_t[b * (2 * H_) + dir * H_ + hc] = h;
    }
    __syncthreads();
  }
}

// --------------------------------- driver ----------------------------------

extern "C" void kernel_launch(void* const* d_in, const int* in_sizes, int n_in,
                              void* d_out, int out_size, void* d_ws, size_t ws_size,
                              hipStream_t stream) {
  (void)in_sizes; (void)n_in; (void)out_size; (void)ws_size;

  const float* x      = (const float*)d_in[0];
  const float* l1_W   = (const float*)d_in[1];
  const float* l1_U   = (const float*)d_in[2];
  const float* l1_b   = (const float*)d_in[3];
  const float* l1_Wb  = (const float*)d_in[4];
  const float* l1_Ub  = (const float*)d_in[5];
  const float* l1_bb  = (const float*)d_in[6];
  const float* l2_Wx  = (const float*)d_in[7];
  const float* l2_Wh  = (const float*)d_in[8];
  const float* l2_U   = (const float*)d_in[9];
  const float* l2_b   = (const float*)d_in[10];
  const float* l2_Wxb = (const float*)d_in[11];
  const float* l2_Whb = (const float*)d_in[12];
  const float* l2_Ub  = (const float*)d_in[13];
  const float* l2_bb  = (const float*)d_in[14];

  char* ws = (char*)d_ws;
  size_t off = 0;
  auto carve = [&](size_t bytes) -> char* {
    char* p = ws + off;
    off += (bytes + 255) & ~(size_t)255;
    return p;
  };
  unsigned short* xT  = (unsigned short*)carve((size_t)T_ * B_ * D_ * 2);
  unsigned short* W1f = (unsigned short*)carve((size_t)NG * D_ * 2);
  unsigned short* W1b = (unsigned short*)carve((size_t)NG * D_ * 2);
  unsigned short* U1f = (unsigned short*)carve((size_t)NG * H_ * 2);
  unsigned short* U1b = (unsigned short*)carve((size_t)NG * H_ * 2);
  unsigned short* W2f = (unsigned short*)carve((size_t)NG * 512 * 2);
  unsigned short* W2b = (unsigned short*)carve((size_t)NG * 512 * 2);
  unsigned short* U2f = (unsigned short*)carve((size_t)H_ * H_ * 2);
  unsigned short* U2b = (unsigned short*)carve((size_t)H_ * H_ * 2);
  unsigned short* hpf = (unsigned short*)carve((size_t)T_ * B_ * H_ * 2);
  unsigned short* hpb = (unsigned short*)carve((size_t)T_ * B_ * H_ * 2);
  float* xpF = (float*)carve((size_t)T_ * B_ * NG * 4);
  float* xpB = (float*)carve((size_t)T_ * B_ * NG * 4);

  // 1) layout conversions (f32 -> bf16, B-fragment friendly layouts)
  k_conv_x<<<(T_ * B_ * D_) / 256, 256, 0, stream>>>(x, xT);
  k_conv_w<<<(NG * D_) / 256, 256, 0, stream>>>(l1_W, W1f, D_);
  k_conv_w<<<(NG * D_) / 256, 256, 0, stream>>>(l1_Wb, W1b, D_);
  k_conv_w<<<(NG * H_) / 256, 256, 0, stream>>>(l1_U, U1f, H_);
  k_conv_w<<<(NG * H_) / 256, 256, 0, stream>>>(l1_Ub, U1b, H_);
  k_conv_w2<<<(NG * 512) / 256, 256, 0, stream>>>(l2_Wx, l2_Wh, W2f);
  k_conv_w2<<<(NG * 512) / 256, 256, 0, stream>>>(l2_Wxb, l2_Whb, W2b);
  k_conv_u2<<<(H_ * H_) / 256, 256, 0, stream>>>(l2_U, U2f);
  k_conv_u2<<<(H_ * H_) / 256, 256, 0, stream>>>(l2_Ub, U2b);

  // 2) layer-1 input projections (bias folded in; bwd reads time-reversed x)
  dim3 gg(TBROWS / 128, NG / 128);
  k_gemm<<<gg, 256, 0, stream>>>(xT, nullptr, W1f, l1_b,  xpF, D_, D_, 0);
  k_gemm<<<gg, 256, 0, stream>>>(xT, nullptr, W1b, l1_bb, xpB, D_, D_, 1);

  // 3) layer-1 recurrent scans (fwd + bwd concurrently), TDM-staged xp rows
  size_t smem1 = (size_t)32 * GLD * 4 + (size_t)32 * HPAD * 2
               + (size_t)32 * NG * 4;            // gates + h + staged xp row
  k_scan1<<<2, 1024, smem1, stream>>>(xpF, xpB, U1f, U1b, hpf, hpb);

  // 4) layer-2 projections: x*Wx + hp*Wh as a single K=512 GEMM (reuse xp bufs)
  k_gemm<<<gg, 256, 0, stream>>>(xT, hpf, W2f, nullptr, xpF, D_, 512, 0);
  k_gemm<<<gg, 256, 0, stream>>>(xT, hpb, W2b, nullptr, xpB, D_, 512, 1);

  // 5) layer-2 recurrent scans -> d_out = [y_t2 (B x 2H) | y_net2 (B,T,2H)]
  float* out_t = (float*)d_out;
  float* out_net = out_t + (size_t)B_ * 2 * H_;
  k_scan2<<<2, 512, 0, stream>>>(xpF, xpB, U2f, U2b, l2_b, l2_bb, out_t, out_net);
}
